// ResGCN1TestNodeType_19791209300121
// MI455X (gfx1250) — matrix operand order repaired and checked by reading, not compiled, not run on hardware
//
#include <hip/hip_runtime.h>
#include <hip/hip_bf16.h>

// ---------------------------------------------------------------------------
// ResGCN (3 node types, 6 relations, 5 layers) for MI455X / gfx1250.
//
//   * transform-then-scatter: X = h_src @ W_rel (node-granularity WMMA GEMM,
//     bf16 inputs, fp32 accum), then per-edge fp32 atomic scatter-add.
//   * features stored bf16 (halves HBM/L2 traffic; memory-bound workload),
//     aggregation buffers + residuals fp32.
//   * W_rel pre-packed into the exact per-lane WMMA B-fragment layout so the
//     GEMM inner loop B-load is a single coalesced 32B load per lane.
//   * bf16-A GEMM double-buffers the A tile with GLOBAL_LOAD_ASYNC_TO_LDS_B128
//     (ASYNCcnt + s_wait_asynccnt): the async DMA of tile kt+1 overlaps the
//     WMMAs of tile kt; single barrier per K-step.
// ---------------------------------------------------------------------------

typedef __bf16 bf16_t;
typedef __attribute__((ext_vector_type(16))) __bf16 v16bf;
typedef __attribute__((ext_vector_type(8)))  __bf16 v8bf;
typedef __attribute__((ext_vector_type(8)))  float  v8f;
typedef __attribute__((ext_vector_type(4)))  int    v4i;

#if defined(__HIP_DEVICE_COMPILE__) && \
    __has_builtin(__builtin_amdgcn_global_load_async_to_lds_b128) && \
    __has_builtin(__builtin_amdgcn_s_wait_asynccnt)
#define USE_ASYNC_LDS 1
#else
#define USE_ASYNC_LDS 0
#endif

#define AS1 __attribute__((address_space(1)))
#define AS3 __attribute__((address_space(3)))

__device__ __forceinline__ bf16_t f2bf(float f) { return (bf16_t)f; }

// ---------------------------------------------------------------------------
// Pack a row-major fp32 [K x Ncols] matrix into WMMA B-fragment layout (bf16).
// Fragment (kt, nt): 32 lanes x 16 elems; lane L covers col nt*16+(L&15),
// K = kt*32 + (L<16 ? 0 : 16) + e, the contiguous-K order the A fragments from LDS also use.
// K beyond Kreal zero-filled (pads 300 -> 320 for the encoders).
// ---------------------------------------------------------------------------
__global__ void pack_b_kernel(const float* __restrict__ src, bf16_t* __restrict__ dst,
                              int nmat, size_t srcStride, size_t dstStride,
                              int Kreal, int nkt, int nnt, int Ncols)
{
    int gid  = blockIdx.x * 256 + threadIdx.x;
    int lane = gid & 31;
    int t    = gid >> 5;
    int nt   = t % nnt; t /= nnt;
    int kt   = t % nkt; t /= nkt;
    int mat  = t;
    if (mat >= nmat) return;
    const float* s = src + (size_t)mat * srcStride;
    bf16_t* d = dst + (size_t)mat * dstStride +
                ((size_t)(kt * nnt + nt) * 32 + (size_t)lane) * 16;
    int col   = nt * 16 + (lane & 15);
    int kbase = kt * 32 + ((lane < 16) ? 0 : 16);
#pragma unroll
    for (int e = 0; e < 16; ++e) {
        int k = kbase + e;
        float v = (k < Kreal) ? s[(size_t)k * Ncols + col] : 0.0f;
        d[e] = f2bf(v);
    }
}

// ---------------------------------------------------------------------------
// Shared GEMM geometry: 256 threads = 8 waves; block tile 128(M) x 64(N);
// wave tile 32x32 = 2x2 v8f accumulators. LDS A-tile row stride 40 bf16
// (80B: 16B-aligned ds_load_b128 fragments, conflict-free bank stride).
// Out-of-range rows are staged by clamping to row M-1 (their products are
// never stored), keeping the staging path branch-free and vectorizable.
// ---------------------------------------------------------------------------
#define GEMM_LDSS 40

__device__ __forceinline__ void gemm_compute_step(
    const bf16_t* __restrict__ Abuf, const bf16_t* __restrict__ Bpack,
    int kt, int nkt, int NT, int ntile_base, int lane, int wr, v8f (&acc)[2][2])
{
    // B fragments: pre-packed, one 32B coalesced load per lane
    v16bf bfrag[2];
#pragma unroll
    for (int tn = 0; tn < 2; ++tn) {
        int ntile = ntile_base + tn;
        bfrag[tn] = *reinterpret_cast<const v16bf*>(
            Bpack + ((size_t)(kt * NT + ntile) * 32 + (size_t)lane) * 16);
    }
    // prefetch next K-step's B fragment (lowers to global_prefetch_b8)
    if (kt + 1 < nkt)
        __builtin_prefetch(
            Bpack + ((size_t)((kt + 1) * NT + ntile_base) * 32 + (size_t)lane) * 16,
            0, 3);

    // A fragments from LDS (contiguous K per lane half, matching packed B)
    v16bf afrag[2];
    const int kb = (lane < 16) ? 0 : 16;
#pragma unroll
    for (int tm = 0; tm < 2; ++tm) {
        int r = wr * 32 + tm * 16 + (lane & 15);
        const bf16_t* ap = &Abuf[r * GEMM_LDSS + kb];
        v8bf lo = *reinterpret_cast<const v8bf*>(ap);       // K = kb..kb+7
        v8bf hi = *reinterpret_cast<const v8bf*>(ap + 8);   // K = kb+8..kb+15
        v16bf f;
#pragma unroll
        for (int e = 0; e < 8; ++e) { f[e] = lo[e]; f[e + 8] = hi[e]; }
        afrag[tm] = f;
    }

#pragma unroll
    for (int tm = 0; tm < 2; ++tm)
#pragma unroll
        for (int tn = 0; tn < 2; ++tn)
            acc[tm][tn] = __builtin_amdgcn_wmma_f32_16x16x32_bf16(
                false, afrag[tm], false, bfrag[tn],
                (short)0, acc[tm][tn], false, false);
}

template<bool HAS_BIAS>
__device__ __forceinline__ void gemm_store_tile(
    v8f (&acc)[2][2], float* __restrict__ C, const float* __restrict__ bias,
    int ldc, int M, int m0, int nblk0, int wr, int wc, int lane)
{
    // 16x16 f32 C/D layout: lane = N, vgpr v = M row v (lanes 0-15) / v+8
#pragma unroll
    for (int tm = 0; tm < 2; ++tm) {
#pragma unroll
        for (int tn = 0; tn < 2; ++tn) {
            int col = nblk0 + wc * 32 + tn * 16 + (lane & 15);
#pragma unroll
            for (int v = 0; v < 8; ++v) {
                int row = m0 + wr * 32 + tm * 16 + v + ((lane < 16) ? 0 : 8);
                if (row < M) {
                    float val = acc[tm][tn][v];
                    if (HAS_BIAS) val += bias[col];
                    C[(size_t)row * ldc + col] = val;
                }
            }
        }
    }
}

// ---------------------------------------------------------------------------
// bf16-A WMMA GEMM (main relation transform): C = A(bf16) @ Bpack, fp32 out.
// Double-buffered async A staging: DMA of tile kt+1 overlaps WMMAs of tile kt.
// K must be a multiple of 32 (K = 256 here).
// ---------------------------------------------------------------------------
__global__ __launch_bounds__(256)
void gemm_wmma_bf16_kernel(const bf16_t* __restrict__ A, int lda,
                           const bf16_t* __restrict__ Bpack,
                           float* __restrict__ C, int ldc,
                           int M, int K, int Ncols)
{
    __shared__ bf16_t As[2][128 * GEMM_LDSS];

    const int tid  = threadIdx.x;
    const int lane = tid & 31;
    const int wave = tid >> 5;
    const int wr   = wave & 3;
    const int wc   = wave >> 2;
    const int m0   = blockIdx.x * 128;
    const int NT   = Ncols >> 4;
    const int ntile_base = (blockIdx.y * 64) / 16 + wc * 2;
    const int nkt  = K >> 5;

    v8f zero = {0.f, 0.f, 0.f, 0.f, 0.f, 0.f, 0.f, 0.f};
    v8f acc[2][2];
    acc[0][0] = zero; acc[0][1] = zero; acc[1][0] = zero; acc[1][1] = zero;

    const int loadRow  = tid >> 1;       // 128 rows, 2 threads per row
    const int loadHalf = (tid & 1) * 16; // 16 bf16 (32B) each
    const int arow     = m0 + loadRow;
    const int ar       = (arow < M) ? arow : (M - 1);   // row clamp (see note)
    const bf16_t* srow = A + (size_t)ar * lda + loadHalf;

    // prologue: stage tile 0 into buffer 0
    {
        const bf16_t* src = srow;
        bf16_t* dst = &As[0][loadRow * GEMM_LDSS + loadHalf];
#if USE_ASYNC_LDS
        __builtin_amdgcn_global_load_async_to_lds_b128(
            (AS1 v4i*)const_cast<bf16_t*>(src), (AS3 v4i*)dst, 0, 0);
        __builtin_amdgcn_global_load_async_to_lds_b128(
            (AS1 v4i*)const_cast<bf16_t*>(src + 8), (AS3 v4i*)(dst + 8), 0, 0);
#else
        *(v8bf*)(dst)     = *(const v8bf*)(src);
        *(v8bf*)(dst + 8) = *(const v8bf*)(src + 8);
#endif
    }

    for (int kt = 0; kt < nkt; ++kt) {
#if USE_ASYNC_LDS
        __builtin_amdgcn_s_wait_asynccnt(0);   // own stage-kt DMA done
#endif
        __syncthreads();                        // all waves' stage-kt visible;
                                                // all stage-(kt-1) readers done
        if (kt + 1 < nkt) {                     // overlap: DMA kt+1 vs WMMA kt
            const bf16_t* src = srow + (kt + 1) * 32;
            bf16_t* dst = &As[(kt + 1) & 1][loadRow * GEMM_LDSS + loadHalf];
#if USE_ASYNC_LDS
            __builtin_amdgcn_global_load_async_to_lds_b128(
                (AS1 v4i*)const_cast<bf16_t*>(src), (AS3 v4i*)dst, 0, 0);
            __builtin_amdgcn_global_load_async_to_lds_b128(
                (AS1 v4i*)const_cast<bf16_t*>(src + 8), (AS3 v4i*)(dst + 8), 0, 0);
#else
            *(v8bf*)(dst)     = *(const v8bf*)(src);
            *(v8bf*)(dst + 8) = *(const v8bf*)(src + 8);
#endif
        }
        gemm_compute_step(As[kt & 1], Bpack, kt, nkt, NT, ntile_base, lane, wr, acc);
    }

    gemm_store_tile<false>(acc, C, nullptr, ldc, M, m0, blockIdx.y * 64, wr, wc, lane);
}

// ---------------------------------------------------------------------------
// fp32-A WMMA GEMM (encoders): converts to bf16 while staging; K zero-padded
// to Kpad (multiple of 32). Single-buffer, two barriers per K-step.
// ---------------------------------------------------------------------------
__global__ __launch_bounds__(256)
void gemm_wmma_f32_kernel(const float* __restrict__ A, int lda,
                          const bf16_t* __restrict__ Bpack,
                          const float* __restrict__ bias,
                          float* __restrict__ C, int ldc,
                          int M, int Kpad, int Kreal, int Ncols)
{
    __shared__ bf16_t As[128 * GEMM_LDSS];

    const int tid  = threadIdx.x;
    const int lane = tid & 31;
    const int wave = tid >> 5;
    const int wr   = wave & 3;
    const int wc   = wave >> 2;
    const int m0   = blockIdx.x * 128;
    const int NT   = Ncols >> 4;
    const int ntile_base = (blockIdx.y * 64) / 16 + wc * 2;
    const int nkt  = Kpad >> 5;

    v8f zero = {0.f, 0.f, 0.f, 0.f, 0.f, 0.f, 0.f, 0.f};
    v8f acc[2][2];
    acc[0][0] = zero; acc[0][1] = zero; acc[1][0] = zero; acc[1][1] = zero;

    const int loadRow  = tid >> 1;
    const int loadHalf = (tid & 1) * 16;
    const int arow     = m0 + loadRow;
    const int ar       = (arow < M) ? arow : (M - 1);

    for (int kt = 0; kt < nkt; ++kt) {
        const int k0 = kt * 32 + loadHalf;
        bf16_t* dst = &As[loadRow * GEMM_LDSS + loadHalf];
        const float* src = A + (size_t)ar * lda + k0;
        if (k0 + 16 <= Kreal) {                    // fast interior path
#pragma unroll
            for (int q = 0; q < 2; ++q) {
                float4 f0 = *(const float4*)(src + q * 8 + 0);
                float4 f1 = *(const float4*)(src + q * 8 + 4);
                v8bf t;
                t[0] = f2bf(f0.x); t[1] = f2bf(f0.y);
                t[2] = f2bf(f0.z); t[3] = f2bf(f0.w);
                t[4] = f2bf(f1.x); t[5] = f2bf(f1.y);
                t[6] = f2bf(f1.z); t[7] = f2bf(f1.w);
                *(v8bf*)(dst + q * 8) = t;
            }
        } else {                                   // partial last K-chunk
#pragma unroll
            for (int e = 0; e < 16; ++e)
                dst[e] = f2bf((k0 + e < Kreal) ? src[e] : 0.0f);
        }
        __syncthreads();
        gemm_compute_step(As, Bpack, kt, nkt, NT, ntile_base, lane, wr, acc);
        __syncthreads();
    }

    gemm_store_tile<true>(acc, C, bias, ldc, M, m0, blockIdx.y * 64, wr, wc, lane);
}

// ---------------------------------------------------------------------------
// Per-edge fp32 scatter-add of transformed rows: 4 edges per block, 256 cols.
// ---------------------------------------------------------------------------
__global__ void scatter_add_kernel(const int* __restrict__ edges, int nE,
                                   const float* __restrict__ X,
                                   float* __restrict__ dst)
{
    int c  = threadIdx.x;
    int e0 = blockIdx.x * 4;
#pragma unroll
    for (int i = 0; i < 4; ++i) {
        int e = e0 + i;
        if (e < nE) {
            int s = edges[e];
            int d = edges[nE + e];
            unsafeAtomicAdd(&dst[(size_t)d * 256 + c], X[(size_t)s * 256 + c]);
        }
    }
}

// new[t] = sum of biases of relations targeting this type (broadcast rows).
__global__ void bias_init_kernel(float* __restrict__ dst, const float* __restrict__ bl,
                                 int r0, int r1, int r2, size_t n)
{
    size_t i = (size_t)blockIdx.x * 256 + threadIdx.x;
    if (i >= n) return;
    int c = (int)(i & 255);
    float v = bl[r0 * 256 + c];
    if (r1 >= 0) v += bl[r1 * 256 + c];
    if (r2 >= 0) v += bl[r2 * 256 + c];
    dst[i] = v;
}

// relu, optional residual add / residual capture, bf16 feature write.
__global__ void epilogue_kernel(const float* __restrict__ nw, float* __restrict__ res,
                                bf16_t* __restrict__ hb, size_t n,
                                int add_res, int set_res)
{
    size_t i = (size_t)blockIdx.x * 256 + threadIdx.x;
    if (i >= n) return;
    float v = fmaxf(nw[i], 0.0f);
    if (add_res) v += res[i];
    if (set_res) res[i] = v;
    hb[i] = f2bf(v);
}

// h[:,0:128] = lbl_emb[labels], h[:,128:256] = encoder GEMM result X.
__global__ void assemble_kernel(const int* __restrict__ labels,
                                const float* __restrict__ lbl_emb,
                                const float* __restrict__ X,
                                bf16_t* __restrict__ hb, size_t n)
{
    size_t i = (size_t)blockIdx.x * 256 + threadIdx.x;
    if (i >= n) return;
    int node = (int)(i >> 8), col = (int)(i & 255);
    float v = (col < 128) ? lbl_emb[(size_t)labels[node] * 128 + col]
                          : X[(size_t)node * 128 + (col - 128)];
    hb[i] = f2bf(v);
}

// h_test rows = t_emb broadcast.
__global__ void broadcast_kernel(const float* __restrict__ t_emb,
                                 bf16_t* __restrict__ hb, size_t n)
{
    size_t i = (size_t)blockIdx.x * 256 + threadIdx.x;
    if (i < n) hb[i] = f2bf(t_emb[i & 255]);
}

// ---------------------------------------------------------------------------
// Decoder + softmax: one wave per node, lane-parallel dot over D=256,
// shfl_xor wave32 reduction.
// ---------------------------------------------------------------------------
template<int NOUT>
__global__ void decoder_kernel(const bf16_t* __restrict__ h,
                               const float* __restrict__ W,
                               const float* __restrict__ b,
                               float* __restrict__ logits,
                               float* __restrict__ soft, int N)
{
    int gid  = blockIdx.x * 256 + threadIdx.x;
    int node = gid >> 5;
    int lane = gid & 31;
    if (node >= N) return;
    float acc[NOUT];
#pragma unroll
    for (int o = 0; o < NOUT; ++o) acc[o] = 0.0f;
    const bf16_t* hp = h + (size_t)node * 256;
#pragma unroll
    for (int j = 0; j < 8; ++j) {
        int k = lane + j * 32;
        float hv = (float)hp[k];
#pragma unroll
        for (int o = 0; o < NOUT; ++o) acc[o] += hv * W[k * NOUT + o];
    }
#pragma unroll
    for (int off = 16; off >= 1; off >>= 1)
#pragma unroll
        for (int o = 0; o < NOUT; ++o)
            acc[o] += __shfl_xor(acc[o], off, 32);
    if (lane == 0) {
        float l[NOUT];
#pragma unroll
        for (int o = 0; o < NOUT; ++o) l[o] = acc[o] + b[o];
        float m = l[0];
#pragma unroll
        for (int o = 1; o < NOUT; ++o) m = fmaxf(m, l[o]);
        float ex[NOUT], s = 0.0f;
#pragma unroll
        for (int o = 0; o < NOUT; ++o) { ex[o] = __expf(l[o] - m); s += ex[o]; }
        float inv = 1.0f / s;
#pragma unroll
        for (int o = 0; o < NOUT; ++o) {
            logits[(size_t)node * NOUT + o] = l[o];
            soft  [(size_t)node * NOUT + o] = ex[o] * inv;
        }
    }
}

// ---------------------------------------------------------------------------
static inline unsigned cdiv(size_t a, size_t b) { return (unsigned)((a + b - 1) / b); }

extern "C" void kernel_launch(void* const* d_in, const int* in_sizes, int n_in,
                              void* d_out, int out_size, void* d_ws, size_t ws_size,
                              hipStream_t stream)
{
    (void)n_in; (void)out_size; (void)ws_size;

    const int*   c_labels  = (const int*)  d_in[0];
    const float* c_content = (const float*)d_in[1];
    const int*   a_labels  = (const int*)  d_in[2];
    const float* a_content = (const float*)d_in[3];
    const int*   E[6] = { (const int*)d_in[4], (const int*)d_in[5], (const int*)d_in[6],
                          (const int*)d_in[7], (const int*)d_in[8], (const int*)d_in[9] };
    const float* c_lbl_emb = (const float*)d_in[10];
    const float* c_enc_W   = (const float*)d_in[11];
    const float* c_enc_b   = (const float*)d_in[12];
    const float* a_lbl_emb = (const float*)d_in[13];
    const float* a_enc_W   = (const float*)d_in[14];
    const float* a_enc_b   = (const float*)d_in[15];
    const float* t_emb     = (const float*)d_in[16];
    const float* W_rel     = (const float*)d_in[17];
    const float* b_rel     = (const float*)d_in[18];
    const float* dec_W     = (const float*)d_in[19];
    const float* dec_b     = (const float*)d_in[20];
    const float* adec_W    = (const float*)d_in[21];
    const float* adec_b    = (const float*)d_in[22];

    const int Ncfg  = in_sizes[0];
    const int Nast  = in_sizes[2];
    const int Ntest = 1000;
    const int D = 256;
    int nE[6];
    for (int r = 0; r < 6; ++r) nE[r] = in_sizes[4 + r] / 2;

    // ---- workspace layout ----
    char* ws = (char*)d_ws;
    size_t off = 0;
    auto take = [&](size_t bytes) -> char* {
        char* p = ws + off;
        off = (off + bytes + 255) & ~(size_t)255;
        return p;
    };
    int Nn[3] = { Ncfg, Nast, Ntest };
    bf16_t* hb[3];  float* res[3];  float* nw[3];
    for (int t = 0; t < 3; ++t) hb[t]  = (bf16_t*)take((size_t)Nn[t] * D * sizeof(bf16_t));
    for (int t = 0; t < 3; ++t) res[t] = (float*) take((size_t)Nn[t] * D * sizeof(float));
    for (int t = 0; t < 3; ++t) nw[t]  = (float*) take((size_t)Nn[t] * D * sizeof(float));
    float*  X   = (float*) take((size_t)Nast * D * sizeof(float));
    bf16_t* BpR = (bf16_t*)take((size_t)30 * D * D * sizeof(bf16_t));       // 5 layers x 6 rels
    bf16_t* BpC = (bf16_t*)take((size_t)10 * 8 * 32 * 16 * sizeof(bf16_t)); // K=320, N=128
    bf16_t* BpA = (bf16_t*)take((size_t)10 * 8 * 32 * 16 * sizeof(bf16_t));

    // ---- pack weights into WMMA B-fragment layout ----
    pack_b_kernel<<<cdiv((size_t)30 * 8 * 16 * 32, 256), 256, 0, stream>>>(
        W_rel, BpR, 30, (size_t)D * D, (size_t)D * D, 256, 8, 16, 256);
    pack_b_kernel<<<cdiv((size_t)10 * 8 * 32, 256), 256, 0, stream>>>(
        c_enc_W, BpC, 1, 0, 0, 300, 10, 8, 128);
    pack_b_kernel<<<cdiv((size_t)10 * 8 * 32, 256), 256, 0, stream>>>(
        a_enc_W, BpA, 1, 0, 0, 300, 10, 8, 128);

    // ---- encoders: X = content @ enc_W + b (WMMA, K padded 300->320) ----
    gemm_wmma_f32_kernel<<<dim3(cdiv(Ncfg, 128), 2), 256, 0, stream>>>(
        c_content, 300, BpC, c_enc_b, X, 128, Ncfg, 320, 300, 128);
    assemble_kernel<<<cdiv((size_t)Ncfg * D, 256), 256, 0, stream>>>(
        c_labels, c_lbl_emb, X, hb[0], (size_t)Ncfg * D);
    gemm_wmma_f32_kernel<<<dim3(cdiv(Nast, 128), 2), 256, 0, stream>>>(
        a_content, 300, BpA, a_enc_b, X, 128, Nast, 320, 300, 128);
    assemble_kernel<<<cdiv((size_t)Nast * D, 256), 256, 0, stream>>>(
        a_labels, a_lbl_emb, X, hb[1], (size_t)Nast * D);
    broadcast_kernel<<<cdiv((size_t)Ntest * D, 256), 256, 0, stream>>>(
        t_emb, hb[2], (size_t)Ntest * D);

    // ---- 5 GCN layers ----
    const int relSrc[6] = { 0, 1, 1, 0, 0, 2 };
    const int relDst[6] = { 0, 1, 0, 1, 2, 0 };
    for (int l = 0; l < 5; ++l) {
        const float* bl = b_rel + (size_t)l * 6 * D;
        bias_init_kernel<<<cdiv((size_t)Ncfg  * D, 256), 256, 0, stream>>>(nw[0], bl, 0, 2, 5,  (size_t)Ncfg  * D);
        bias_init_kernel<<<cdiv((size_t)Nast  * D, 256), 256, 0, stream>>>(nw[1], bl, 1, 3, -1, (size_t)Nast  * D);
        bias_init_kernel<<<cdiv((size_t)Ntest * D, 256), 256, 0, stream>>>(nw[2], bl, 4, -1, -1, (size_t)Ntest * D);

        for (int r = 0; r < 6; ++r) {
            int s = relSrc[r], d = relDst[r];
            // X = h[s] @ W_rel[l][r]  (bf16 WMMA, fp32 out, async-pipelined)
            gemm_wmma_bf16_kernel<<<dim3(cdiv(Nn[s], 128), 4), 256, 0, stream>>>(
                hb[s], D, BpR + (size_t)(l * 6 + r) * D * D,
                X, D, Nn[s], 256, 256);
            // new[d] += segment_sum(X[src], dst)
            scatter_add_kernel<<<cdiv(nE[r], 4), 256, 0, stream>>>(E[r], nE[r], X, nw[d]);
        }
        int addr = (l == 1 || l == 3) ? 1 : 0;
        int setr = (l == 0 || l == 2) ? 1 : 0;
        for (int t = 0; t < 3; ++t)
            epilogue_kernel<<<cdiv((size_t)Nn[t] * D, 256), 256, 0, stream>>>(
                nw[t], res[t], hb[t], (size_t)Nn[t] * D, addr, setr);
    }

    // ---- decoders + softmax ----
    float* out = (float*)d_out;
    float* cfg_logits = out;
    float* cfg_soft   = out + (size_t)Ncfg * 2;
    float* ast_logits = out + (size_t)Ncfg * 4;
    float* ast_soft   = out + (size_t)Ncfg * 4 + (size_t)Nast * 3;
    decoder_kernel<2><<<cdiv((size_t)Ncfg * 32, 256), 256, 0, stream>>>(
        hb[0], dec_W, dec_b, cfg_logits, cfg_soft, Ncfg);
    decoder_kernel<3><<<cdiv((size_t)Nast * 32, 256), 256, 0, stream>>>(
        hb[1], adec_W, adec_b, ast_logits, ast_soft, Nast);
}
